// PaiNNInteraction_9758165696761
// MI455X (gfx1250) — compile-verified, matrix operand order checked by reading, not run on hardware
//
#include <hip/hip_runtime.h>

typedef __attribute__((ext_vector_type(16))) _Float16 v16h;
typedef __attribute__((ext_vector_type(8)))  _Float16 v8h;
typedef __attribute__((ext_vector_type(8)))  float    v8f;

#define FDIM 128
#define F3   384
#define LDSROW 136   // 128 + 8 halves pad: 272B row stride, breaks bank conflicts

// ---------------------------------------------------------------------------
// Kernel A: initialize output with q (first N*F floats) and mu (next N*3*F).
// ---------------------------------------------------------------------------
__global__ void painn_init(const float* __restrict__ q,
                           const float* __restrict__ mu,
                           float* __restrict__ out, int nq, int ntot) {
  int i = blockIdx.x * blockDim.x + threadIdx.x;
  if (i < ntot) {
    out[i] = (i < nq) ? q[i] : mu[i - nq];
  }
}

// ---------------------------------------------------------------------------
// Kernel P: pack a f32 weight matrix [FDIM, ncols] into WMMA B-fragment order
// as f16:  Wp[tile][kchunk][lane][e]  (tile = 16 cols, kchunk = 32 K-values).
// Per-lane fragment becomes one contiguous 32-byte load in the GEMM.
// ---------------------------------------------------------------------------
__global__ void painn_pack(const float* __restrict__ W, _Float16* __restrict__ Wp,
                           int ncols, int total) {
  int t = blockIdx.x * blockDim.x + threadIdx.x;
  if (t >= total) return;
  const int e     = t & 15;
  const int lane  = (t >> 4) & 31;
  const int chunk = (t >> 9) & 3;        // FDIM/32 = 4 K-chunks
  const int tile  = t >> 11;             // 2048 halves per (tile)
  const int m     = lane & 15;
  const int kb    = (lane >> 4) * 8;
  const int k     = chunk * 32 + kb + (e & 7) + ((e >> 3) << 4);
  const int n     = tile * 16 + m;
  Wp[t] = (_Float16)W[k * ncols + n];
}

// ---------------------------------------------------------------------------
// Kernel B: ctx = silu(q @ W1 + b1) @ W2 + b2   (WMMA f16 -> f32 accumulate)
// One block = 16 rows. 8 waves. q tile staged in LDS as f16 (coalesced),
// A-fragments read via 16B LDS loads, B-fragments via packed 32B loads.
// ---------------------------------------------------------------------------
__global__ __launch_bounds__(256) void painn_ctx(
    const float* __restrict__ q,
    const v16h* __restrict__ W1p, const float* __restrict__ b1,
    const v16h* __restrict__ W2p, const float* __restrict__ b2,
    float* __restrict__ ctx) {
  __shared__ _Float16 qh[16 * LDSROW];
  __shared__ _Float16 hbuf[16 * LDSROW];

  const int lane    = threadIdx.x & 31;
  const int wave    = threadIdx.x >> 5;   // 0..7
  const int row0    = blockIdx.x * 16;    // N = 20000 divisible by 16
  const int m       = lane & 15;
  const int kb      = (lane >> 4) * 8;    // K sub-offset per 16-bit A layout
  const int half_hi = lane >> 4;          // 0/1 -> C/D row offset

  // Stage q tile -> LDS as f16 (coalesced global reads).
  for (int idx = threadIdx.x; idx < 16 * FDIM; idx += 256) {
    const int r = idx >> 7, c = idx & 127;
    qh[r * LDSROW + c] = (_Float16)q[(size_t)(row0 + r) * FDIM + c];
  }
  __syncthreads();

  // ---- Layer 1: h = silu(q @ W1 + b1); wave w -> column tile w ----
  {
    const int n0 = wave * 16;
    v8f acc = {};
#pragma unroll
    for (int kc = 0; kc < 4; ++kc) {      // 4 chunks of K=32
      const v8h a_lo = *(const v8h*)&qh[m * LDSROW + kc * 32 + kb];
      const v8h a_hi = *(const v8h*)&qh[m * LDSROW + kc * 32 + kb + 16];
      v16h a;
#pragma unroll
      for (int e = 0; e < 8; ++e) { a[e] = a_lo[e]; a[8 + e] = a_hi[e]; }
      const v16h b = W1p[(wave * 4 + kc) * 32 + lane];
      acc = __builtin_amdgcn_wmma_f32_16x16x32_f16(
          false, a, false, b, (short)0, acc, false, false);
    }
#pragma unroll
    for (int v = 0; v < 8; ++v) {
      const int mm  = v + half_hi * 8;
      const int nn  = n0 + m;
      const float x = acc[v] + b1[nn];
      // SiLU via hw reciprocal: x * rcp(1 + exp(-x))
      const float s = x * __builtin_amdgcn_rcpf(1.0f + __expf(-x));
      hbuf[mm * LDSROW + nn] = (_Float16)s;
    }
  }
  __syncthreads();

  // ---- Layer 2: ctx = h @ W2 + b2; wave w -> tiles w, w+8, w+16 ----
  for (int t = wave; t < 24; t += 8) {    // uniform 3 iterations per wave
    const int c0 = t * 16;
    v8f acc = {};
#pragma unroll
    for (int kc = 0; kc < 4; ++kc) {
      const v8h a_lo = *(const v8h*)&hbuf[m * LDSROW + kc * 32 + kb];
      const v8h a_hi = *(const v8h*)&hbuf[m * LDSROW + kc * 32 + kb + 16];
      v16h a;
#pragma unroll
      for (int e = 0; e < 8; ++e) { a[e] = a_lo[e]; a[8 + e] = a_hi[e]; }
      const v16h b = W2p[(t * 4 + kc) * 32 + lane];
      acc = __builtin_amdgcn_wmma_f32_16x16x32_f16(
          false, a, false, b, (short)0, acc, false, false);
    }
#pragma unroll
    for (int v = 0; v < 8; ++v) {
      const int mm = v + half_hi * 8;
      const int nn = c0 + m;
      ctx[(size_t)(row0 + mm) * F3 + nn] = acc[v] + b2[nn];
    }
  }
}

// ---------------------------------------------------------------------------
// Kernel C: fused edge phase. 128 threads per edge (one per feature f).
// W_ij is streamed once -> non-temporal loads keep L2 free for ctx/mu/out.
// Scatter-adds use f32 no-return atomics (STOREcnt path).
// ---------------------------------------------------------------------------
__global__ __launch_bounds__(256) void painn_edges(
    const float* __restrict__ Wij, const float* __restrict__ dir,
    const float* __restrict__ mu,  const float* __restrict__ ctx,
    const int* __restrict__ idx_i, const int* __restrict__ idx_j,
    float* __restrict__ out_q, float* __restrict__ out_mu, int E) {
  const long long t = (long long)blockIdx.x * blockDim.x + threadIdx.x;
  const int e = (int)(t >> 7);
  const int f = (int)(t & 127);
  if (e >= E) return;

  const int j = idx_j[e];
  const int i = idx_i[e];

  const float cq = ctx[(size_t)j * F3 + f];
  const float cr = ctx[(size_t)j * F3 + FDIM + f];
  const float cm = ctx[(size_t)j * F3 + 2 * FDIM + f];
  const float wq = __builtin_nontemporal_load(&Wij[(size_t)e * F3 + f]);
  const float wr = __builtin_nontemporal_load(&Wij[(size_t)e * F3 + FDIM + f]);
  const float wm = __builtin_nontemporal_load(&Wij[(size_t)e * F3 + 2 * FDIM + f]);

  const float dmr = wr * cr;
  const float dmm = wm * cm;
  const float d0 = dir[e * 3 + 0];
  const float d1 = dir[e * 3 + 1];
  const float d2 = dir[e * 3 + 2];

  const float* muj  = mu     + (size_t)j * 3 * FDIM + f;
  float*       outi = out_mu + (size_t)i * 3 * FDIM + f;

  __hip_atomic_fetch_add(&out_q[(size_t)i * FDIM + f], wq * cq,
                         __ATOMIC_RELAXED, __HIP_MEMORY_SCOPE_AGENT);
  __hip_atomic_fetch_add(outi,            dmr * d0 + dmm * muj[0],
                         __ATOMIC_RELAXED, __HIP_MEMORY_SCOPE_AGENT);
  __hip_atomic_fetch_add(outi + FDIM,     dmr * d1 + dmm * muj[FDIM],
                         __ATOMIC_RELAXED, __HIP_MEMORY_SCOPE_AGENT);
  __hip_atomic_fetch_add(outi + 2 * FDIM, dmr * d2 + dmm * muj[2 * FDIM],
                         __ATOMIC_RELAXED, __HIP_MEMORY_SCOPE_AGENT);
}

// ---------------------------------------------------------------------------
extern "C" void kernel_launch(void* const* d_in, const int* in_sizes, int n_in,
                              void* d_out, int out_size, void* d_ws, size_t ws_size,
                              hipStream_t stream) {
  const float* q   = (const float*)d_in[0];   // [N, F]
  const float* mu  = (const float*)d_in[1];   // [N, 3, F]
  const float* Wij = (const float*)d_in[2];   // [E, 3F]
  const float* dir = (const float*)d_in[3];   // [E, 3]
  const float* W1  = (const float*)d_in[4];   // [F, F]
  const float* b1  = (const float*)d_in[5];   // [F]
  const float* W2  = (const float*)d_in[6];   // [F, 3F]
  const float* b2  = (const float*)d_in[7];   // [3F]
  const int* idx_i = (const int*)d_in[8];     // [E]
  const int* idx_j = (const int*)d_in[9];     // [E]

  const int N = in_sizes[0] / FDIM;
  const int E = in_sizes[3] / 3;

  float* out    = (float*)d_out;
  float* out_q  = out;                        // [N, F]
  float* out_mu = out + (size_t)N * FDIM;     // [N, 3, F]

  // Workspace layout: ctx [N*3F floats], then packed f16 weights.
  float*    ctx    = (float*)d_ws;
  const size_t ctx_bytes = (size_t)N * F3 * sizeof(float);   // 30.72 MB (32B-aligned)
  _Float16* W1p = (_Float16*)((char*)d_ws + ctx_bytes);      // 8 tiles * 2048 halves
  const int w1p_total = (FDIM / 16) * 4 * 32 * 16;           // 16384
  _Float16* W2p = W1p + w1p_total;                           // 24 tiles * 2048 halves
  const int w2p_total = (F3 / 16) * 4 * 32 * 16;             // 49152

  const int nq   = N * FDIM;
  const int ntot = nq + N * 3 * FDIM;

  painn_init<<<(ntot + 255) / 256, 256, 0, stream>>>(q, mu, out, nq, ntot);

  painn_pack<<<(w1p_total + 255) / 256, 256, 0, stream>>>(W1, W1p, FDIM, w1p_total);
  painn_pack<<<(w2p_total + 255) / 256, 256, 0, stream>>>(W2, W2p, F3,   w2p_total);

  painn_ctx<<<(N + 15) / 16, 256, 0, stream>>>(
      q, (const v16h*)W1p, b1, (const v16h*)W2p, b2, ctx);

  const long long edge_threads = (long long)E * FDIM;
  painn_edges<<<(unsigned)((edge_threads + 255) / 256), 256, 0, stream>>>(
      Wij, dir, mu, ctx, idx_i, idx_j, out_q, out_mu, E);
}
